// Attention_Coupled_29987461661022
// MI455X (gfx1250) — matrix operand order repaired and checked by reading, not compiled, not run on hardware
//
#include <hip/hip_runtime.h>

#define B_ 64
#define S_ 4096
#define D_ 256
#define BN_EPS 1e-5f

// LDS layout (bytes)
#define W1_STRIDE_B 528              // 264 bf16 per row
#define W1_BYTES    (256 * W1_STRIDE_B)   // 135168
#define XS_STRIDE_B 1040             // 260 f32 per row
#define XS_BYTES    (128 * XS_STRIDE_B)   // 133120
#define SMEM_BYTES  (W1_BYTES + XS_BYTES) // 268288

// WMMA fragment types (CDNA5 wave32)
typedef __attribute__((ext_vector_type(16))) __bf16 vbf16;
typedef __attribute__((ext_vector_type(8)))  float  v8f;

union FragB { vbf16 v; uint4 q[2]; };
union FragA { vbf16 v; unsigned u[8]; };

// f32 pair -> packed bf16x2 (prefer hardware v_cvt_pk_bf16_f32)
#if __has_builtin(__builtin_amdgcn_cvt_pk_bf16_f32)
__device__ inline unsigned f2bf_pk(float a, float b) {
  auto t = __builtin_amdgcn_cvt_pk_bf16_f32(a, b);
  unsigned u;
  __builtin_memcpy(&u, &t, sizeof(u));
  return u;
}
#else
__device__ inline unsigned f2bf_pk(float a, float b) {
  union { float f; unsigned u; } x, y; x.f = a; y.f = b;
  unsigned lo = (x.u + 0x7FFFu + ((x.u >> 16) & 1u)) >> 16;
  unsigned hi = (y.u + 0x7FFFu + ((y.u >> 16) & 1u)) >> 16;
  return lo | (hi << 16);
}
#endif

// hardware tanh (CDNA5 TRANS op). Builtin if available, else raw v_tanh_f32.
#if __has_builtin(__builtin_amdgcn_tanhf)
__device__ inline float fast_tanh(float x) { return __builtin_amdgcn_tanhf(x); }
#else
__device__ inline float fast_tanh(float x) {
  float r;
  // v_nop covers the TRANS-op result-use hazard (hardware does not interlock)
  asm volatile("v_tanh_f32 %0, %1\n\tv_nop" : "=v"(r) : "v"(x));
  return r;
}
#endif

// issue 32 async b128 global->LDS copies for one 128x256 f32 tile (per thread)
__device__ inline void async_stage_tile(const float* __restrict__ gbase,
                                        unsigned ldsBase, int tid) {
  #pragma unroll 8
  for (int j = 0; j < 32; ++j) {
    int i = tid + j * 256;                 // float4 index within tile
    int row = i >> 6;                      // 64 float4 per 256-wide row
    unsigned lds  = ldsBase + (unsigned)(row * XS_STRIDE_B) + ((i & 63) << 4);
    unsigned goff = (unsigned)i << 4;      // byte offset from tile base
    asm volatile("global_load_async_to_lds_b128 %0, %1, %2"
                 :: "v"(lds), "v"(goff), "s"(gbase)
                 : "memory");
  }
}

__device__ inline void wait_async0() {
  asm volatile("s_wait_asynccnt 0" ::: "memory");
}

// ---------------------------------------------------------------------------
// Kernel 1: fused fc1 (bf16 WMMA) + BatchNorm(eval) + tanh + dot(x_h) -> a_t
// grid (8, 64), block 256 (8 waves), dynamic LDS = SMEM_BYTES.
// X tiles arrive via async global->LDS DMA, prefetched one s-block ahead.
// ---------------------------------------------------------------------------
__global__ __launch_bounds__(256)
void k_fc1_attn(const float* __restrict__ xh,     // [B, D]
                const float* __restrict__ xp,     // [B, S, D]
                const float* __restrict__ W1,     // [D, D] (e, d) row-major
                const float* __restrict__ gamma,  // [S]
                const float* __restrict__ beta,   // [S]
                const float* __restrict__ rmean,  // [S]
                const float* __restrict__ rvar,   // [S]
                float* __restrict__ a_t)          // [B, S]
{
  extern __shared__ char smem[];
  char* W1s = smem;                          // bf16, stride 264
  char* Xs  = smem + W1_BYTES;               // f32,  stride 260
  const unsigned xsLds = (unsigned)(uintptr_t)Xs;  // LDS byte offset

  const int tid  = threadIdx.x;
  const int b    = blockIdx.y;
  const int wave = tid >> 5;
  const int lane = tid & 31;
  const int m    = lane & 15;   // N / row-within-16 index
  const int hf   = lane >> 4;   // lane half

  const float* xbase = xp + ((size_t)b * S_ + (size_t)blockIdx.x * 512) * D_;

  // prefetch s-block 0 while we stage W1
  async_stage_tile(xbase, xsLds, tid);

  // ---- stage W1 (f32 -> bf16) once per workgroup ----
  for (int i = tid; i < (D_ * D_) / 4; i += 256) {
    float4 f = reinterpret_cast<const float4*>(W1)[i];
    int row = i >> 6;
    int col = (i & 63) << 2;
    uint2 p;
    p.x = f2bf_pk(f.x, f.y);
    p.y = f2bf_pk(f.z, f.w);
    *reinterpret_cast<uint2*>(W1s + row * W1_STRIDE_B + col * 2) = p;
  }

  // x_h slice this lane needs: xhv[nt] multiplies column N = nt*16 + m
  float xhv[16];
  #pragma unroll
  for (int nt = 0; nt < 16; ++nt) xhv[nt] = xh[b * D_ + nt * 16 + m];

  for (int sb = 0; sb < 4; ++sb) {
    const int sBase = (blockIdx.x * 4 + sb) * 128;

    // BN affine for rows handled by (VGPR r, half hf): M = r + 8*hf
    float scale[8], shift[8];
    #pragma unroll
    for (int r = 0; r < 8; ++r) {
      int s = sBase + wave * 16 + hf * 8 + r;
      float sc = gamma[s] * rsqrtf(rvar[s] + BN_EPS);
      scale[r] = sc;
      shift[r] = beta[s] - rmean[s] * sc;
    }

    wait_async0();        // this wave's DMA for tile sb complete
    __syncthreads();      // everyone's DMA complete (and W1 staged on sb==0)

    // ---- fill A fragments (f32 LDS -> bf16 regs); only read of Xs ----
    // A 16x32 layout (matches B): lane holds row m; lanes<16: K 0..15,
    // lanes>=16: K 16..31.
    FragA afr[8];
    const char* arow = Xs + (wave * 16 + m) * XS_STRIDE_B;
    #pragma unroll
    for (int kt = 0; kt < 8; ++kt) {
      const char* p0 = arow + kt * 128 + hf * 64;        // K 16hf+0..7
      const char* p1 = p0 + 32;                          // K 16hf+8..15
      float4 a0 = *reinterpret_cast<const float4*>(p0);
      float4 a1 = *reinterpret_cast<const float4*>(p0 + 16);
      float4 a2 = *reinterpret_cast<const float4*>(p1);
      float4 a3 = *reinterpret_cast<const float4*>(p1 + 16);
      afr[kt].u[0] = f2bf_pk(a0.x, a0.y);
      afr[kt].u[1] = f2bf_pk(a0.z, a0.w);
      afr[kt].u[2] = f2bf_pk(a1.x, a1.y);
      afr[kt].u[3] = f2bf_pk(a1.z, a1.w);
      afr[kt].u[4] = f2bf_pk(a2.x, a2.y);
      afr[kt].u[5] = f2bf_pk(a2.z, a2.w);
      afr[kt].u[6] = f2bf_pk(a3.x, a3.y);
      afr[kt].u[7] = f2bf_pk(a3.z, a3.w);
    }
    __syncthreads();      // all waves done reading Xs

    // prefetch next s-block; DMA overlaps the whole compute phase below
    if (sb < 3) async_stage_tile(xbase + (size_t)(sb + 1) * 128 * D_, xsLds, tid);

    float dacc[8];
    #pragma unroll
    for (int r = 0; r < 8; ++r) dacc[r] = 0.f;

    // ---- 16 N-tiles, K=256 each via 8 x v_wmma_f32_16x16x32_bf16 ----
    for (int nt = 0; nt < 16; ++nt) {
      // B 32x16 layout: lane holds column n=m of W1 row (nt*16+m);
      // lanes<16: K 0..15, lanes>=16: K 16..31 -> one 32B contiguous chunk.
      FragB bfr[8];
      const char* brow = W1s + (nt * 16 + m) * W1_STRIDE_B + hf * 32;
      #pragma unroll
      for (int kt = 0; kt < 8; ++kt) {
        bfr[kt].q[0] = *reinterpret_cast<const uint4*>(brow + kt * 64);
        bfr[kt].q[1] = *reinterpret_cast<const uint4*>(brow + kt * 64 + 16);
      }
      v8f c = {};
      #pragma unroll
      for (int kt = 0; kt < 8; ++kt) {
        c = __builtin_amdgcn_wmma_f32_16x16x32_bf16(
                false, afr[kt].v, false, bfr[kt].v, (short)0, c, false, false);
      }
      // epilogue: BN affine + tanh + partial dot with x_h
      #pragma unroll
      for (int r = 0; r < 8; ++r) {
        float hb = c[r] * scale[r] + shift[r];
        dacc[r] += fast_tanh(hb) * xhv[nt];
      }
    }

    // reduce dot across the 16 lanes holding one row (N = 0..15)
    #pragma unroll
    for (int r = 0; r < 8; ++r) {
      float v = dacc[r];
      v += __shfl_xor(v, 8, 16);
      v += __shfl_xor(v, 4, 16);
      v += __shfl_xor(v, 2, 16);
      v += __shfl_xor(v, 1, 16);
      dacc[r] = v;
    }
    if (m == 0) {
      #pragma unroll
      for (int r = 0; r < 8; ++r)
        a_t[(size_t)b * S_ + sBase + wave * 16 + hf * 8 + r] = dacc[r];
    }
  }
}

// ---------------------------------------------------------------------------
// Kernel 2: softmax over S per batch. grid 64, block 256.
// ---------------------------------------------------------------------------
__global__ __launch_bounds__(256)
void k_softmax(const float* __restrict__ a_t, float* __restrict__ a_s) {
  __shared__ float red[256];
  const int b = blockIdx.x, tid = threadIdx.x;
  const float* in = a_t + (size_t)b * S_;
  float* out = a_s + (size_t)b * S_;

  float lmax = -3.4e38f;
  for (int i = tid; i < S_; i += 256) lmax = fmaxf(lmax, in[i]);
  red[tid] = lmax; __syncthreads();
  for (int o = 128; o > 0; o >>= 1) {
    if (tid < o) red[tid] = fmaxf(red[tid], red[tid + o]);
    __syncthreads();
  }
  const float gmax = red[0];
  __syncthreads();

  float lsum = 0.f;
  for (int i = tid; i < S_; i += 256) {
    float e = __expf(in[i] - gmax);
    out[i] = e;
    lsum += e;
  }
  red[tid] = lsum; __syncthreads();
  for (int o = 128; o > 0; o >>= 1) {
    if (tid < o) red[tid] += red[tid + o];
    __syncthreads();
  }
  const float inv = 1.f / red[0];
  __syncthreads();
  for (int i = tid; i < S_; i += 256) out[i] *= inv;
}

// ---------------------------------------------------------------------------
// Kernel 3: partial[b,ch,d] = sum over 512-row chunk of a_s * x_hpre.
// grid (8, 64), block 256 (thread = d). Pure streaming bandwidth.
// ---------------------------------------------------------------------------
__global__ __launch_bounds__(256)
void k_wsum(const float* __restrict__ xp, const float* __restrict__ a_s,
            float* __restrict__ partial) {
  __shared__ float as_s[512];
  const int b = blockIdx.y, ch = blockIdx.x, tid = threadIdx.x;
  for (int i = tid; i < 512; i += 256)
    as_s[i] = a_s[(size_t)b * S_ + ch * 512 + i];
  __syncthreads();

  const float* x = xp + ((size_t)b * S_ + ch * 512) * D_ + tid;
  float acc0 = 0.f, acc1 = 0.f, acc2 = 0.f, acc3 = 0.f;
  #pragma unroll 2
  for (int s = 0; s < 512; s += 4) {
    acc0 += as_s[s]     * x[(size_t)s * D_];
    acc1 += as_s[s + 1] * x[(size_t)(s + 1) * D_];
    acc2 += as_s[s + 2] * x[(size_t)(s + 2) * D_];
    acc3 += as_s[s + 3] * x[(size_t)(s + 3) * D_];
  }
  partial[((size_t)b * 8 + ch) * D_ + tid] = (acc0 + acc1) + (acc2 + acc3);
}

// ---------------------------------------------------------------------------
// Kernel 4: out[b,d] = W2[d] * sum_ch partial[b,ch,d]. grid 64, block 256.
// ---------------------------------------------------------------------------
__global__ __launch_bounds__(256)
void k_final(const float* __restrict__ partial, const float* __restrict__ W2,
             float* __restrict__ out) {
  const int b = blockIdx.x, d = threadIdx.x;
  float acc = 0.f;
  #pragma unroll
  for (int c = 0; c < 8; ++c) acc += partial[((size_t)b * 8 + c) * D_ + d];
  out[(size_t)b * D_ + d] = acc * W2[d];
}

extern "C" void kernel_launch(void* const* d_in, const int* in_sizes, int n_in,
                              void* d_out, int out_size, void* d_ws, size_t ws_size,
                              hipStream_t stream) {
  const float* xh    = (const float*)d_in[0];  // [B,1,D]
  const float* xp    = (const float*)d_in[1];  // [B,S,D]
  const float* W1    = (const float*)d_in[2];  // [D,D]
  const float* W2    = (const float*)d_in[3];  // [D,1]
  const float* gamma = (const float*)d_in[4];  // [S]
  const float* beta  = (const float*)d_in[5];  // [S]
  const float* rmean = (const float*)d_in[6];  // [S]
  const float* rvar  = (const float*)d_in[7];  // [S]
  float* out = (float*)d_out;

  float* a_t     = (float*)d_ws;                 // B*S
  float* a_s     = a_t + (size_t)B_ * S_;        // B*S
  float* partial = a_s + (size_t)B_ * S_;        // B*8*D

  k_fc1_attn<<<dim3(8, 64), 256, SMEM_BYTES, stream>>>(xh, xp, W1, gamma, beta,
                                                       rmean, rvar, a_t);
  k_softmax<<<64, 256, 0, stream>>>(a_t, a_s);
  k_wsum<<<dim3(8, 64), 256, 0, stream>>>(xp, a_s, partial);
  k_final<<<64, 256, 0, stream>>>(partial, W2, out);
}
